// EdgeConv_16174846837133
// MI455X (gfx1250) — compile-verified, matrix operand order checked by reading, not run on hardware
//
#include <hip/hip_runtime.h>

typedef __attribute__((ext_vector_type(16))) __bf16 v16bf;
typedef __attribute__((ext_vector_type(16))) float  v16f;
typedef __attribute__((ext_vector_type(8)))  float  v8f;

#define HID       64
#define NDIM      64
#define EDIM      16

// packed weight layout in ws (bf16 element indices)
#define W1F_OFF   0        // [5][4][32][16]  = 10240
#define W2F_OFF   10240    // [2][4][32][16]  =  4096
#define WUF_OFF   14336    // [4][4][32][16]  =  8192
#define PACK_TOT  22528
#define AGG_BYTE_OFF 49152 // aligned start of float agg[N*64], then counts[N]

union FragBF {
    v16bf v;
    uint4 q[2];
};

__device__ __forceinline__ void lf8(const float* p, float* v) {
    const float4* q = (const float4*)p;
    float4 a = q[0], b = q[1];
    v[0]=a.x; v[1]=a.y; v[2]=a.z; v[3]=a.w;
    v[4]=b.x; v[5]=b.y; v[6]=b.z; v[7]=b.w;
}

// ---------------- weight packing: B-fragment layout -----------------------
// lane<16: n=nt*16+lane,  elems 0..7 -> k=kt*32+0..7,  elems 8..15 -> k=kt*32+16..23
// lane>=16: n=nt*16+lane-16, elems 0..7 -> k=kt*32+8..15, 8..15 -> kt*32+24..31
__global__ void pack_weights(const float* __restrict__ W1,
                             const float* __restrict__ W2,
                             const float* __restrict__ Wu,
                             __bf16* __restrict__ wpack) {
    int tid = blockIdx.x * blockDim.x + threadIdx.x;
    if (tid >= PACK_TOT) return;
    int j, kmax; const float* W;
    if (tid < W2F_OFF)      { j = tid;            W = W1; kmax = 2*NDIM + EDIM; } // 144
    else if (tid < WUF_OFF) { j = tid - W2F_OFF;  W = W2; kmax = HID;           } // 64
    else                    { j = tid - WUF_OFF;  W = Wu; kmax = NDIM + HID;    } // 128
    int elem  = j & 15;
    int lane  = (j >> 4) & 31;
    int ntile = (j >> 9) & 3;
    int ktile = j >> 11;
    int n = ntile * 16 + (lane & 15);
    int k = ktile * 32 + ((lane >= 16) ? 8 : 0) + (elem & 7) + ((elem >= 8) ? 16 : 0);
    float val = (k < kmax) ? W[k * HID + n] : 0.0f;
    wpack[tid] = (__bf16)val;
}

__global__ void zero_ws(float* __restrict__ p, long long n) {
    long long i = (long long)blockIdx.x * blockDim.x + threadIdx.x;
    long long step = (long long)gridDim.x * blockDim.x;
    for (; i < n; i += step) p[i] = 0.0f;
}

// ---------------- edge message kernel -------------------------------------
__global__ void __launch_bounds__(256)
edge_kernel(const float* __restrict__ x,
            const int*   __restrict__ srcIdx,
            const int*   __restrict__ dstIdx,
            const float* __restrict__ ea,
            const float* __restrict__ b1g,
            const float* __restrict__ b2g,
            const __bf16* __restrict__ wpack,
            float* __restrict__ agg,
            float* __restrict__ cnt,
            int E) {
    __shared__ __attribute__((aligned(16))) __bf16 sW1[5][4][32][16]; // 20KB
    __shared__ __attribute__((aligned(16))) __bf16 sW2[2][4][32][16]; //  8KB
    __shared__ __attribute__((aligned(16))) __bf16 sH[8][16 * HID];   // 16KB
    __shared__ float sB1[HID], sB2[HID];
    __shared__ int   sDst[8][16];

    int tid = threadIdx.x;
    // stage packed weights to LDS
    for (int i = tid; i < 5120; i += 256) ((unsigned*)sW1)[i] = ((const unsigned*)(wpack + W1F_OFF))[i];
    for (int i = tid; i < 2048; i += 256) ((unsigned*)sW2)[i] = ((const unsigned*)(wpack + W2F_OFF))[i];
    if (tid < HID) { sB1[tid] = b1g[tid]; sB2[tid] = b2g[tid]; }
    __syncthreads();

    int lane  = tid & 31;
    int col16 = lane & 15;
    bool hi   = lane >= 16;
    int w     = tid >> 5;
    int wave  = blockIdx.x * 8 + w;
    int nWav  = gridDim.x * 8;
    int T     = (E + 15) >> 4;

    for (int tile = wave; tile < T; tile += nWav) {
        int e = tile * 16 + col16;
        bool valid = e < E;
        int s = valid ? srcIdx[e] : 0;
        int d = valid ? dstIdx[e] : 0;
        if (!hi) {
            sDst[w][col16] = d;
            if (valid) atomicAdd(&cnt[d], 1.0f);
        }
        // prefetch next tile's edge_attr rows
        int nt2 = tile + nWav;
        if (nt2 < T) __builtin_prefetch(ea + (size_t)(nt2 * 16 + col16) * EDIM, 0, 0);

        // ---- build A fragments for [x[src] | x[dst] | attr | 0] (K=160) ----
        FragBF a1[5];
        #pragma unroll
        for (int kt = 0; kt < 5; ++kt) {
            int c0 = kt * 32 + (hi ? 8 : 0);
            float v[16];
            #pragma unroll
            for (int half = 0; half < 2; ++half) {
                int c = c0 + half * 16;
                float* vh = v + half * 8;
                if (valid && c < NDIM) {
                    lf8(x + (size_t)s * NDIM + c, vh);
                } else if (valid && c < 2 * NDIM) {
                    lf8(x + (size_t)d * NDIM + (c - NDIM), vh);
                } else if (valid && c < 2 * NDIM + EDIM) {
                    lf8(ea + (size_t)e * EDIM + (c - 2 * NDIM), vh);
                } else {
                    for (int i = 0; i < 8; ++i) vh[i] = 0.0f;
                }
            }
            v16f fv;
            #pragma unroll
            for (int i = 0; i < 16; ++i) fv[i] = v[i];
            a1[kt].v = __builtin_convertvector(fv, v16bf);   // native bf16 cvt
        }

        // ---- layer 1: H = relu(in @ W1 + b1), write to per-wave LDS pad ----
        #pragma unroll
        for (int nt = 0; nt < 4; ++nt) {
            v8f c = {};
            #pragma unroll
            for (int kt = 0; kt < 5; ++kt) {
                FragBF b;
                const uint4* p = (const uint4*)&sW1[kt][nt][lane][0];
                b.q[0] = p[0]; b.q[1] = p[1];
                c = __builtin_amdgcn_wmma_f32_16x16x32_bf16(false, a1[kt].v, false, b.v,
                                                            (short)0, c, false, false);
            }
            float bias = sB1[nt * 16 + col16];
            #pragma unroll
            for (int r = 0; r < 8; ++r) {
                float h = c[r] + bias;
                h = h > 0.0f ? h : 0.0f;
                int m = r + (hi ? 8 : 0);
                sH[w][m * HID + nt * 16 + col16] = (__bf16)h;
            }
        }
        asm volatile("s_wait_dscnt 0" ::: "memory");

        // ---- re-layout H (C-layout -> A-layout) via LDS ----
        FragBF a2[2];
        #pragma unroll
        for (int kt = 0; kt < 2; ++kt) {
            int c0 = kt * 32 + (hi ? 8 : 0);
            a2[kt].q[0] = *(const uint4*)&sH[w][col16 * HID + c0];
            a2[kt].q[1] = *(const uint4*)&sH[w][col16 * HID + c0 + 16];
        }

        // ---- layer 2 + atomic scatter-add ----
        #pragma unroll
        for (int nt = 0; nt < 4; ++nt) {
            v8f c = {};
            #pragma unroll
            for (int kt = 0; kt < 2; ++kt) {
                FragBF b;
                const uint4* p = (const uint4*)&sW2[kt][nt][lane][0];
                b.q[0] = p[0]; b.q[1] = p[1];
                c = __builtin_amdgcn_wmma_f32_16x16x32_bf16(false, a2[kt].v, false, b.v,
                                                            (short)0, c, false, false);
            }
            float bias = sB2[nt * 16 + col16];
            #pragma unroll
            for (int r = 0; r < 8; ++r) {
                int m = r + (hi ? 8 : 0);
                int ee = tile * 16 + m;
                if (ee < E) {
                    int dd = sDst[w][m];
                    atomicAdd(&agg[(size_t)dd * HID + nt * 16 + col16], c[r] + bias);
                }
            }
        }
    }
}

// ---------------- node update kernel --------------------------------------
__global__ void __launch_bounds__(256)
node_kernel(const float* __restrict__ x,
            const __bf16* __restrict__ wupack,
            const float* __restrict__ bug,
            const float* __restrict__ agg,
            const float* __restrict__ cnt,
            float* __restrict__ out,
            int N) {
    __shared__ __attribute__((aligned(16))) __bf16 sWu[4][4][32][16]; // 16KB
    __shared__ float sBu[NDIM];
    int tid = threadIdx.x;
    for (int i = tid; i < 4096; i += 256) ((unsigned*)sWu)[i] = ((const unsigned*)wupack)[i];
    if (tid < NDIM) sBu[tid] = bug[tid];
    __syncthreads();

    int lane  = tid & 31;
    int col16 = lane & 15;
    bool hi   = lane >= 16;
    int wave  = blockIdx.x * 8 + (tid >> 5);
    int nWav  = gridDim.x * 8;
    int T     = (N + 15) >> 4;

    for (int tile = wave; tile < T; tile += nWav) {
        int v = tile * 16 + col16;
        bool valid = v < N;
        float c0f = valid ? cnt[v] : 0.0f;
        float inv = 1.0f / (c0f + 1e-6f);

        FragBF a[4];
        #pragma unroll
        for (int kt = 0; kt < 4; ++kt) {
            int cb = kt * 32 + (hi ? 8 : 0);
            float vv[16];
            #pragma unroll
            for (int half = 0; half < 2; ++half) {
                int c = cb + half * 16;
                float* vh = vv + half * 8;
                if (valid && c < NDIM) {
                    lf8(x + (size_t)v * NDIM + c, vh);
                } else if (valid) {
                    lf8(agg + (size_t)v * HID + (c - NDIM), vh);
                    #pragma unroll
                    for (int i = 0; i < 8; ++i) vh[i] *= inv;
                } else {
                    for (int i = 0; i < 8; ++i) vh[i] = 0.0f;
                }
            }
            v16f fv;
            #pragma unroll
            for (int i = 0; i < 16; ++i) fv[i] = vv[i];
            a[kt].v = __builtin_convertvector(fv, v16bf);
        }

        #pragma unroll
        for (int nt = 0; nt < 4; ++nt) {
            v8f c = {};
            #pragma unroll
            for (int kt = 0; kt < 4; ++kt) {
                FragBF b;
                const uint4* p = (const uint4*)&sWu[kt][nt][lane][0];
                b.q[0] = p[0]; b.q[1] = p[1];
                c = __builtin_amdgcn_wmma_f32_16x16x32_bf16(false, a[kt].v, false, b.v,
                                                            (short)0, c, false, false);
            }
            float bias = sBu[nt * 16 + col16];
            #pragma unroll
            for (int r = 0; r < 8; ++r) {
                int m = r + (hi ? 8 : 0);
                int node = tile * 16 + m;
                if (node < N) {
                    size_t o = (size_t)node * NDIM + nt * 16 + col16;
                    float val = c[r] + bias;
                    val = val > 0.0f ? val : 0.0f;
                    out[o] = x[o] + val;
                }
            }
        }
    }
}

extern "C" void kernel_launch(void* const* d_in, const int* in_sizes, int n_in,
                              void* d_out, int out_size, void* d_ws, size_t ws_size,
                              hipStream_t stream) {
    const float* x  = (const float*)d_in[0];
    const int*   ei = (const int*)d_in[1];
    const float* ea = (const float*)d_in[2];
    const float* W1 = (const float*)d_in[3];
    const float* b1 = (const float*)d_in[4];
    const float* W2 = (const float*)d_in[5];
    const float* b2 = (const float*)d_in[6];
    const float* Wu = (const float*)d_in[7];
    const float* bu = (const float*)d_in[8];
    float* out = (float*)d_out;

    int N = in_sizes[0] / NDIM;   // 100000
    int E = in_sizes[1] / 2;      // 1600000

    char* ws = (char*)d_ws;
    __bf16* wpack = (__bf16*)ws;
    float* agg = (float*)(ws + AGG_BYTE_OFF);
    float* cnt = agg + (size_t)N * HID;

    // 1) pack weights into WMMA B-fragment layout (bf16)
    pack_weights<<<(PACK_TOT + 255) / 256, 256, 0, stream>>>(W1, W2, Wu, wpack);
    // 2) zero agg + counts
    long long nz = (long long)N * HID + N;
    zero_ws<<<2048, 256, 0, stream>>>(agg, nz);
    // 3) edge messages + scatter
    edge_kernel<<<1024, 256, 0, stream>>>(x, ei, ei + E, ea, b1, b2, wpack, agg, cnt, E);
    // 4) node update + residual
    node_kernel<<<512, 256, 0, stream>>>(x, wpack + WUF_OFF, bu, agg, cnt, out, N);
}
